// LSTM_40673340293566
// MI455X (gfx1250) — compile-verified, matrix operand order checked by reading, not run on hardware
//
#include <hip/hip_runtime.h>
#include <hip/hip_bf16.h>

// ---------------------------------------------------------------------------
// Problem constants (match reference setup_inputs)
// ---------------------------------------------------------------------------
#define BB   4096
#define TT   5
#define DD   1024
#define HH   512
#define CC   20
#define BT   (BB * TT)       // 20480
#define G4H  (4 * HH)        // 2048
#define EPSV 1e-5f

typedef __attribute__((ext_vector_type(16))) __bf16 v16bf;
typedef __attribute__((ext_vector_type(8)))  float  v8f;

union FragBF {
    v16bf v;
    uint4 q[2];
};

__device__ __forceinline__ unsigned short f2bf(float f) {
    unsigned int u = __float_as_uint(f);
    unsigned int r = u + 0x7FFFu + ((u >> 16) & 1u);  // round-to-nearest-even
    return (unsigned short)(r >> 16);
}

__device__ __forceinline__ float sigm(float x) {
    return 1.0f / (1.0f + __expf(-x));
}

// ---------------------------------------------------------------------------
// Zero-init buffer (h, c start at zero; d_ws is poisoned)
// ---------------------------------------------------------------------------
__global__ void k_zero(float* __restrict__ p, int n) {
    int i = blockIdx.x * blockDim.x + threadIdx.x;
    if (i < n) p[i] = 0.0f;
}

// ---------------------------------------------------------------------------
// f32 -> bf16 elementwise (weight conversion, done every call for determinism)
// ---------------------------------------------------------------------------
__global__ void k_f32_to_bf16(const float* __restrict__ in,
                              unsigned short* __restrict__ out, int n) {
    int i = blockIdx.x * blockDim.x + threadIdx.x;
    if (i < n) out[i] = f2bf(in[i]);
}

// ---------------------------------------------------------------------------
// BN1: training-mode batchnorm over rows of x [BT, DD], one block per column.
// Writes normalized bf16 for the WMMA GEMM.
// ---------------------------------------------------------------------------
__global__ void k_bn_input(const float* __restrict__ x,
                           const float* __restrict__ gamma,
                           const float* __restrict__ beta,
                           unsigned short* __restrict__ xbn) {
    const int d   = blockIdx.x;      // column
    const int tid = threadIdx.x;     // 256 threads
    __shared__ float s0[256], s1[256];
    __shared__ float stat[2];

    float sum = 0.f, sq = 0.f;
    for (int r = tid; r < BT; r += 256) {
        float v = x[(size_t)r * DD + d];
        sum += v; sq += v * v;
    }
    s0[tid] = sum; s1[tid] = sq;
    __syncthreads();
    for (int off = 128; off > 0; off >>= 1) {
        if (tid < off) { s0[tid] += s0[tid + off]; s1[tid] += s1[tid + off]; }
        __syncthreads();
    }
    if (tid == 0) {
        float m = s0[0] * (1.0f / BT);
        float v = s1[0] * (1.0f / BT) - m * m;
        stat[0] = m;
        stat[1] = rsqrtf(v + EPSV);
    }
    __syncthreads();
    const float m  = stat[0];
    const float gi = gamma[d] * stat[1];
    const float bt = beta[d];
    for (int r = tid; r < BT; r += 256) {
        float v = x[(size_t)r * DD + d];
        xbn[(size_t)r * DD + d] = f2bf((v - m) * gi + bt);
    }
}

// ---------------------------------------------------------------------------
// BN on h and c each step. One block per hidden column j.
// h -> bf16 h_bn (GEMM input).  c -> normalized in place (fp32, cell input).
// ---------------------------------------------------------------------------
__global__ void k_bn_hc(const float* __restrict__ h,
                        float* __restrict__ c,
                        unsigned short* __restrict__ hbn,
                        const float* __restrict__ gamma,
                        const float* __restrict__ beta) {
    const int j   = blockIdx.x;
    const int tid = threadIdx.x;
    __shared__ float s0[256], s1[256];
    __shared__ float stat[2];

    // ---- h statistics + normalize to bf16 ----
    float sum = 0.f, sq = 0.f;
    for (int r = tid; r < BB; r += 256) {
        float v = h[(size_t)r * HH + j];
        sum += v; sq += v * v;
    }
    s0[tid] = sum; s1[tid] = sq;
    __syncthreads();
    for (int off = 128; off > 0; off >>= 1) {
        if (tid < off) { s0[tid] += s0[tid + off]; s1[tid] += s1[tid + off]; }
        __syncthreads();
    }
    if (tid == 0) {
        float m = s0[0] * (1.0f / BB);
        float v = s1[0] * (1.0f / BB) - m * m;
        stat[0] = m;
        stat[1] = rsqrtf(v + EPSV);
    }
    __syncthreads();
    {
        const float m  = stat[0];
        const float gi = gamma[j] * stat[1];
        const float bt = beta[j];
        for (int r = tid; r < BB; r += 256) {
            float v = h[(size_t)r * HH + j];
            hbn[(size_t)r * HH + j] = f2bf((v - m) * gi + bt);
        }
    }
    __syncthreads();   // done reading stat[] before reuse

    // ---- c statistics + normalize in place ----
    sum = 0.f; sq = 0.f;
    for (int r = tid; r < BB; r += 256) {
        float v = c[(size_t)r * HH + j];
        sum += v; sq += v * v;
    }
    s0[tid] = sum; s1[tid] = sq;
    __syncthreads();
    for (int off = 128; off > 0; off >>= 1) {
        if (tid < off) { s0[tid] += s0[tid + off]; s1[tid] += s1[tid + off]; }
        __syncthreads();
    }
    if (tid == 0) {
        float m = s0[0] * (1.0f / BB);
        float v = s1[0] * (1.0f / BB) - m * m;
        stat[0] = m;
        stat[1] = rsqrtf(v + EPSV);
    }
    __syncthreads();
    {
        const float m  = stat[0];
        const float gi = gamma[j] * stat[1];
        const float bt = beta[j];
        for (int r = tid; r < BB; r += 256) {
            float v = c[(size_t)r * HH + j];
            c[(size_t)r * HH + j] = (v - m) * gi + bt;
        }
    }
}

// ---------------------------------------------------------------------------
// bf16 WMMA GEMM:  C[m,n] (+)= sum_k A[m,k] * W[n,k]   (W row-major [N,K])
// A row m lives at A + m*aRowStride + aBase (lets us address x[:,t,:] slices).
// accumulate==0: init accumulator with bias0[n]+bias1[n]; ==1: load C.
//
// Register blocking: each wave owns a 32(M) x 64(N) region = 2x4 WMMA tiles.
// Per K-step of 32: 6 fragment loads (12 x b128) feed 8 WMMAs, with each A
// fragment reused 4x and each B fragment reused 2x in registers.
// Block = 8 waves (2x4) -> 64(M) x 256(N) per block.
// ---------------------------------------------------------------------------
__global__ __launch_bounds__(256)
void k_gemm_bf16(const unsigned short* __restrict__ A, int aRowStride, int aBase,
                 const unsigned short* __restrict__ W, int K,
                 float* __restrict__ C, int ldc,
                 const float* __restrict__ bias0, const float* __restrict__ bias1,
                 int accumulate) {
    const int lane = threadIdx.x & 31;
    const int wave = threadIdx.x >> 5;
    const int wm   = wave >> 2;                   // 0..1
    const int wn   = wave & 3;                    // 0..3
    const int m0   = blockIdx.x * 64 + wm * 32;   // wave M range: m0..m0+31
    const int n0   = blockIdx.y * 256 + wn * 64;  // wave N range: n0..n0+63
    const int half = lane >> 4;        // K half selector per §7.12.2 bf16 layout
    const int r    = lane & 15;

    // Per-lane fragment source rows (A rows / W rows are both K-contiguous).
    const unsigned short* aRow0 = A + (size_t)(m0 + r)      * aRowStride + aBase;
    const unsigned short* aRow1 = A + (size_t)(m0 + 16 + r) * aRowStride + aBase;
    const unsigned short* wRow0 = W + (size_t)(n0 + r)      * K;
    const unsigned short* wRow1 = W + (size_t)(n0 + 16 + r) * K;
    const unsigned short* wRow2 = W + (size_t)(n0 + 32 + r) * K;
    const unsigned short* wRow3 = W + (size_t)(n0 + 48 + r) * K;

    v8f acc[2][4];
    if (accumulate) {
#pragma unroll
        for (int mi = 0; mi < 2; ++mi)
#pragma unroll
            for (int nj = 0; nj < 4; ++nj)
#pragma unroll
                for (int i = 0; i < 8; ++i)
                    acc[mi][nj][i] =
                        C[(size_t)(m0 + mi * 16 + half * 8 + i) * ldc
                          + (n0 + nj * 16 + r)];
    } else {
#pragma unroll
        for (int nj = 0; nj < 4; ++nj) {
            const int n = n0 + nj * 16 + r;
            const float bv = bias0[n] + bias1[n];
#pragma unroll
            for (int mi = 0; mi < 2; ++mi)
#pragma unroll
                for (int i = 0; i < 8; ++i)
                    acc[mi][nj][i] = bv;
        }
    }

    const int ko = half * 8;   // lanes 0-15: K [0..7],[16..23]; 16-31: [8..15],[24..31]
    for (int k0 = 0; k0 < K; k0 += 32) {
        FragBF a0, a1, b0, b1, b2, b3;
        a0.q[0] = *(const uint4*)(aRow0 + k0 + ko);
        a0.q[1] = *(const uint4*)(aRow0 + k0 + 16 + ko);
        a1.q[0] = *(const uint4*)(aRow1 + k0 + ko);
        a1.q[1] = *(const uint4*)(aRow1 + k0 + 16 + ko);
        b0.q[0] = *(const uint4*)(wRow0 + k0 + ko);
        b0.q[1] = *(const uint4*)(wRow0 + k0 + 16 + ko);
        b1.q[0] = *(const uint4*)(wRow1 + k0 + ko);
        b1.q[1] = *(const uint4*)(wRow1 + k0 + 16 + ko);
        b2.q[0] = *(const uint4*)(wRow2 + k0 + ko);
        b2.q[1] = *(const uint4*)(wRow2 + k0 + 16 + ko);
        b3.q[0] = *(const uint4*)(wRow3 + k0 + ko);
        b3.q[1] = *(const uint4*)(wRow3 + k0 + 16 + ko);

        acc[0][0] = __builtin_amdgcn_wmma_f32_16x16x32_bf16(
            false, a0.v, false, b0.v, (short)0, acc[0][0], false, false);
        acc[0][1] = __builtin_amdgcn_wmma_f32_16x16x32_bf16(
            false, a0.v, false, b1.v, (short)0, acc[0][1], false, false);
        acc[0][2] = __builtin_amdgcn_wmma_f32_16x16x32_bf16(
            false, a0.v, false, b2.v, (short)0, acc[0][2], false, false);
        acc[0][3] = __builtin_amdgcn_wmma_f32_16x16x32_bf16(
            false, a0.v, false, b3.v, (short)0, acc[0][3], false, false);
        acc[1][0] = __builtin_amdgcn_wmma_f32_16x16x32_bf16(
            false, a1.v, false, b0.v, (short)0, acc[1][0], false, false);
        acc[1][1] = __builtin_amdgcn_wmma_f32_16x16x32_bf16(
            false, a1.v, false, b1.v, (short)0, acc[1][1], false, false);
        acc[1][2] = __builtin_amdgcn_wmma_f32_16x16x32_bf16(
            false, a1.v, false, b2.v, (short)0, acc[1][2], false, false);
        acc[1][3] = __builtin_amdgcn_wmma_f32_16x16x32_bf16(
            false, a1.v, false, b3.v, (short)0, acc[1][3], false, false);
    }

#pragma unroll
    for (int mi = 0; mi < 2; ++mi)
#pragma unroll
        for (int nj = 0; nj < 4; ++nj)
#pragma unroll
            for (int i = 0; i < 8; ++i)
                C[(size_t)(m0 + mi * 16 + half * 8 + i) * ldc
                  + (n0 + nj * 16 + r)] = acc[mi][nj][i];
}

// ---------------------------------------------------------------------------
// Fused LSTM cell pointwise (PyTorch gate order i,f,g,o) + h_sum accumulation
// c already holds BN(c); h is overwritten with the new hidden state.
// ---------------------------------------------------------------------------
__global__ void k_lstm_cell(const float* __restrict__ gates,
                            float* __restrict__ c,
                            float* __restrict__ h,
                            float* __restrict__ hsum,
                            int first) {
    int idx = blockIdx.x * blockDim.x + threadIdx.x;
    if (idx >= BB * HH) return;
    const int b = idx / HH;
    const int j = idx - b * HH;
    const float* g = gates + (size_t)b * G4H;

    const float ig = sigm(g[j]);
    const float fg = sigm(g[HH + j]);
    const float gg = tanhf(g[2 * HH + j]);
    const float og = sigm(g[3 * HH + j]);

    const float cn = fg * c[idx] + ig * gg;
    const float hn = og * tanhf(cn);
    c[idx] = cn;
    h[idx] = hn;
    hsum[idx] = first ? hn : (hsum[idx] + hn);
}

// ---------------------------------------------------------------------------
// Classifier head: out = (hsum / T) @ W_cls.T + b_cls    (mean of logits)
// One thread per (batch, class) pair, 32-lane groups share a batch row.
// ---------------------------------------------------------------------------
__global__ void k_classifier(const float* __restrict__ hsum,
                             const float* __restrict__ Wc,
                             const float* __restrict__ bc,
                             float* __restrict__ out) {
    int idx = blockIdx.x * blockDim.x + threadIdx.x;
    int b   = idx >> 5;
    int cls = idx & 31;
    if (b >= BB || cls >= CC) return;
    const float* hr = hsum + (size_t)b * HH;
    const float* wr = Wc + (size_t)cls * HH;
    float s = 0.f;
#pragma unroll 4
    for (int j = 0; j < HH; ++j) s = fmaf(hr[j], wr[j], s);
    out[(size_t)b * CC + cls] = s * (1.0f / TT) + bc[cls];
}

// ---------------------------------------------------------------------------
// Launcher
// ---------------------------------------------------------------------------
extern "C" void kernel_launch(void* const* d_in, const int* in_sizes, int n_in,
                              void* d_out, int out_size, void* d_ws, size_t ws_size,
                              hipStream_t stream) {
    const float* features  = (const float*)d_in[0];
    const float* bn1_gamma = (const float*)d_in[1];
    const float* bn1_beta  = (const float*)d_in[2];
    const float* bn2_gamma = (const float*)d_in[3];
    const float* bn2_beta  = (const float*)d_in[4];
    const float* W_ih      = (const float*)d_in[5];
    const float* W_hh      = (const float*)d_in[6];
    const float* b_ih      = (const float*)d_in[7];
    const float* b_hh      = (const float*)d_in[8];
    const float* W_cls     = (const float*)d_in[9];
    const float* b_cls     = (const float*)d_in[10];
    float* out = (float*)d_out;

    // ---- workspace layout (256B aligned chunks) ----
    char* w = (char*)d_ws;
    size_t off = 0;
    auto carve = [&](size_t bytes) {
        char* p = w + off;
        off += (bytes + 255) & ~(size_t)255;
        return p;
    };
    unsigned short* xbn  = (unsigned short*)carve((size_t)BT * DD * 2);   // 41.9 MB
    unsigned short* wih  = (unsigned short*)carve((size_t)G4H * DD * 2);  //  4.2 MB
    unsigned short* whh  = (unsigned short*)carve((size_t)G4H * HH * 2);  //  2.1 MB
    unsigned short* hbn  = (unsigned short*)carve((size_t)BB * HH * 2);   //  4.2 MB
    float* gates = (float*)carve((size_t)BB * G4H * 4);                   // 33.6 MB
    float* h     = (float*)carve((size_t)BB * HH * 4);                    //  8.4 MB
    float* c     = (float*)carve((size_t)BB * HH * 4);                    //  8.4 MB
    float* hsum  = (float*)carve((size_t)BB * HH * 4);                    //  8.4 MB
    (void)ws_size;

    // ---- init h = c = 0 ----
    k_zero<<<(BB * HH + 255) / 256, 256, 0, stream>>>(h, BB * HH);
    k_zero<<<(BB * HH + 255) / 256, 256, 0, stream>>>(c, BB * HH);

    // ---- input batchnorm -> bf16 ----
    k_bn_input<<<DD, 256, 0, stream>>>(features, bn1_gamma, bn1_beta, xbn);

    // ---- weights -> bf16 ----
    k_f32_to_bf16<<<(G4H * DD + 255) / 256, 256, 0, stream>>>(W_ih, wih, G4H * DD);
    k_f32_to_bf16<<<(G4H * HH + 255) / 256, 256, 0, stream>>>(W_hh, whh, G4H * HH);

    const dim3 gemmGrid(BB / 64, G4H / 256);   // 64 x 8 blocks, 256 thr each

    for (int t = 0; t < TT; ++t) {
        // BN(h) -> bf16, BN(c) in place
        k_bn_hc<<<HH, 256, 0, stream>>>(h, c, hbn, bn2_gamma, bn2_beta);

        // gates = x_t @ W_ih.T + (b_ih + b_hh)
        k_gemm_bf16<<<gemmGrid, 256, 0, stream>>>(
            xbn, TT * DD, t * DD, wih, DD, gates, G4H, b_ih, b_hh, /*acc=*/0);

        // gates += h_bn @ W_hh.T
        k_gemm_bf16<<<gemmGrid, 256, 0, stream>>>(
            hbn, HH, 0, whh, HH, gates, G4H, b_ih, b_hh, /*acc=*/1);

        // pointwise cell update + running sum of h
        k_lstm_cell<<<(BB * HH + 255) / 256, 256, 0, stream>>>(
            gates, c, h, hsum, /*first=*/(t == 0) ? 1 : 0);
    }

    // out = mean_t(h_t) @ W_cls.T + b_cls  (head is affine, so mean commutes)
    k_classifier<<<(BB * 32 + 255) / 256, 256, 0, stream>>>(hsum, W_cls, b_cls, out);
}
